// WeldRNN_36541581755031
// MI455X (gfx1250) — matrix operand order repaired and checked by reading, not compiled, not run on hardware
//
#include <hip/hip_runtime.h>

#define HID   1024
#define BATCH 64
#define SEQ   512
#define NIN   6
#define NBLK  64          // one block per 16-column slice of W_hh

typedef __attribute__((ext_vector_type(16))) __bf16   v16bf;
typedef __attribute__((ext_vector_type(8)))  float    v8f;
typedef __attribute__((ext_vector_type(4)))  unsigned v4u;

union AU { v4u u[2]; v16bf v; };

__device__ __forceinline__ unsigned short f2bf(float f) {
    unsigned u = __float_as_uint(f);
    u += 0x7FFFu + ((u >> 16) & 1u);          // round-to-nearest-even
    return (unsigned short)(u >> 16);
}
__device__ __forceinline__ float bf2f(unsigned short h) {
    return __uint_as_float(((unsigned)h) << 16);
}

// ---- init: zero h0 (bf16) and the barrier counter -------------------------
__global__ void init_ws(unsigned short* __restrict__ h0, unsigned* __restrict__ cnt) {
    int i = blockIdx.x * 256 + threadIdx.x;
    h0[i] = 0;
    if (i == 0) *cnt = 0;
}

// ---- persistent RNN: all 512 steps in one kernel --------------------------
// grid = 64 blocks x 256 threads (8 waves). Block b owns output columns
// [16b, 16b+16). Waves: slot=wave>>1 -> batch tile (4x16=64 rows),
// khalf=wave&1 -> K split [0,512)/[512,1024), reduced through LDS.
__global__ __launch_bounds__(256)
void rnn_persist(const float* __restrict__ src,
                 const float* __restrict__ W_ih,
                 const float* __restrict__ b_ih,
                 const float* __restrict__ b_hh,
                 const float* __restrict__ W_hh,
                 const float* __restrict__ W_out,
                 const float* __restrict__ b_out,
                 unsigned short* __restrict__ h0,   // bf16 [B][H]
                 unsigned short* __restrict__ h1,   // bf16 [B][H]
                 float* __restrict__ out,           // f32 [B][S]
                 unsigned* __restrict__ cnt)
{
    // B operand in WMMA register layout: piece p (halves p*8..p*8+7) of the
    // 32B each lane feeds to v_wmma, per k-chunk. 16B per lane per piece ->
    // conflict-free ds_load_b128. 32 KB total.
    __shared__ __align__(16) unsigned short Blds[2][32][32][8];
    __shared__ float red[4][256];

    const int tid  = threadIdx.x;
    const int wave = tid >> 5;
    const int lane = tid & 31;
    const int l16  = lane & 15;
    const int kg   = lane >> 4;
    const int n0   = blockIdx.x * 16;

    // ---- stage this block's 16 rows of W_hh into LDS (f32 -> bf16, swizzled)
    for (int idx = tid; idx < 16 * HID; idx += 256) {
        const int n = idx >> 10;                    // 0..15  (row within slice)
        const int k = idx & (HID - 1);              // coalesced along k
        const unsigned short v = f2bf(W_hh[(n0 + n) * HID + k]);
        const int chunk = k >> 5;
        const int ko    = k & 31;
        const int bkg   = (ko >> 3) & 1;            // which 16-lane group
        const int j     = (ko & 7) | ((ko >> 4) << 3); // half index within lane
        Blds[j >> 3][chunk][n + 16 * bkg][j & 7] = v;
    }

    // ---- hoisted input-projection constants (n fixed per lane forever) ----
    const int nmy = n0 + l16;
    float wih[NIN];
#pragma unroll
    for (int i = 0; i < NIN; ++i) wih[i] = W_ih[nmy * NIN + i];
    const float bias = b_ih[nmy] + b_hh[nmy];

    const int slot  = wave >> 1;
    const int khalf = wave & 1;
    const int m0    = slot * 16;
    // all per-lane invariant address components folded into one offset:
    // A row (M = m0 + lane%16), this wave's K half, and the lane-group's +kg*8
    const int aoff  = (m0 + l16) * HID + khalf * (HID / 2) + kg * 8;

    __syncthreads();

    unsigned short* hc = h0;
    unsigned short* hn = h1;

    for (int t = 0; t < SEQ; ++t) {
        // ---- output projection for step t-1 (block 0 only) ---------------
        if (blockIdx.x == 0 && t > 0) {
            for (int r = 0; r < 8; ++r) {
                int b = wave * 8 + r;
                float p = 0.f;
                for (int n = lane; n < HID; n += 32)
                    p += bf2f(hc[b * HID + n]) * W_out[n];
                for (int off = 16; off > 0; off >>= 1)
                    p += __shfl_xor(p, off, 32);
                if (lane == 0) out[b * SEQ + (t - 1)] = p + b_out[0];
            }
        }

        // ---- C init: fold x_t @ W_ih^T + biases in (khalf==0 wave) -------
        v8f c = {};
        if (khalf == 0) {
#pragma unroll
            for (int v = 0; v < 8; ++v) {
                const int m = m0 + v + 8 * kg;      // C layout: M = vgpr+8*kg
                const float* s = src + (m * SEQ + t) * NIN;
                float acc = bias;
#pragma unroll
                for (int i = 0; i < NIN; ++i) acc += s[i] * wih[i];
                c[v] = acc;
            }
        }

        // ---- WMMA K loop: A from L2 (h), B from LDS ----------------------
        // Fully unrolled: every load is base + compile-time immediate.
        const unsigned short* abase = hc + aoff;
        const int cbase = khalf * 16;
#pragma unroll
        for (int ch = 0; ch < 16; ++ch) {
            AU a, b;
            a.u[0] = *(const v4u*)(abase + ch * 32);        // halves 0-7
            a.u[1] = *(const v4u*)(abase + ch * 32 + 16);   // halves 8-15
            b.u[0] = *(const v4u*)&Blds[0][cbase + ch][lane][0];
            b.u[1] = *(const v4u*)&Blds[1][cbase + ch][lane][0];
            c = __builtin_amdgcn_wmma_f32_16x16x32_bf16(false, a.v, false, b.v,
                                                        (short)0, c, false, false);
        }

        // ---- reduce K halves, relu, store h_{t+1} ------------------------
        if (khalf == 1) {
#pragma unroll
            for (int v = 0; v < 8; ++v) red[slot][v * 32 + lane] = c[v];
        }
        __syncthreads();
        if (khalf == 0) {
#pragma unroll
            for (int v = 0; v < 8; ++v) {
                float x = c[v] + red[slot][v * 32 + lane];
                x = x > 0.f ? x : 0.f;
                hn[(m0 + v + 8 * kg) * HID + nmy] = f2bf(x);
            }
        }

        // ---- device-wide barrier (monotonic counter, deterministic) ------
        __threadfence();                 // release: flush hn stores (global_wb)
        __syncthreads();                 // all threads of block fenced
        if (tid == 0) {
            __hip_atomic_fetch_add(cnt, 1u, __ATOMIC_RELEASE,
                                   __HIP_MEMORY_SCOPE_AGENT);
            const unsigned target = (unsigned)(NBLK * (t + 1));
            while (__hip_atomic_load(cnt, __ATOMIC_RELAXED,
                                     __HIP_MEMORY_SCOPE_AGENT) < target)
                __builtin_amdgcn_s_sleep(2);
            __threadfence();             // acquire: global_inv WGP$ for block
        }
        __syncthreads();

        unsigned short* tmp = hc; hc = hn; hn = tmp;
    }

    // ---- final output column t = SEQ-1 (block 0) --------------------------
    if (blockIdx.x == 0) {
        for (int r = 0; r < 8; ++r) {
            int b = wave * 8 + r;
            float p = 0.f;
            for (int n = lane; n < HID; n += 32)
                p += bf2f(hc[b * HID + n]) * W_out[n];
            for (int off = 16; off > 0; off >>= 1)
                p += __shfl_xor(p, off, 32);
            if (lane == 0) out[b * SEQ + (SEQ - 1)] = p + b_out[0];
        }
    }
}

extern "C" void kernel_launch(void* const* d_in, const int* in_sizes, int n_in,
                              void* d_out, int out_size, void* d_ws, size_t ws_size,
                              hipStream_t stream) {
    const float* src   = (const float*)d_in[0];
    const float* W_ih  = (const float*)d_in[1];
    const float* W_hh  = (const float*)d_in[2];
    const float* b_ih  = (const float*)d_in[3];
    const float* b_hh  = (const float*)d_in[4];
    const float* W_out = (const float*)d_in[5];
    const float* b_out = (const float*)d_in[6];
    float* out = (float*)d_out;

    unsigned short* h0 = (unsigned short*)d_ws;          // 128 KB
    unsigned short* h1 = h0 + BATCH * HID;               // 128 KB
    unsigned*       cn = (unsigned*)(h1 + BATCH * HID);  // 4 B

    init_ws<<<BATCH * HID / 256, 256, 0, stream>>>(h0, cn);
    rnn_persist<<<NBLK, 256, 0, stream>>>(src, W_ih, b_ih, b_hh, W_hh,
                                          W_out, b_out, h0, h1, out, cn);
}
